// HybridEmbedding_21036749816435
// MI455X (gfx1250) — matrix operand order repaired and checked by reading, not compiled, not run on hardware
//
#include <hip/hip_runtime.h>
#include <hip/hip_bf16.h>

typedef _Float16 v16h __attribute__((ext_vector_type(16)));
typedef _Float16 v8h  __attribute__((ext_vector_type(8)));
typedef _Float16 v4h  __attribute__((ext_vector_type(4)));
typedef _Float16 v2h  __attribute__((ext_vector_type(2)));
typedef float    v8f  __attribute__((ext_vector_type(8)));

#define TOK_PER_WAVE    16
#define WAVES_PER_BLOCK 8
#define TOK_PER_BLOCK   128
#define BPACK_BYTES     (128 * 1024)          // 256x256 f16, WMMA-B-fragment order
#define WF4_BYTES       (128 * 16 * 2)        // W_four transposed to [c][j], f16
#define W2T_BYTES       (64 * 32 * 2)         // W2 transposed to [c][j], f16
#define STAGE_BYTES     (BPACK_BYTES + WF4_BYTES + W2T_BYTES)

static __device__ __forceinline__ float clip100(float v) {
    return fminf(fmaxf(v, -100.f), 100.f);
}

static __device__ __forceinline__ float hsum8(v8h a) {
    v4h b = __builtin_shufflevector(a, a, 0, 1, 2, 3) +
            __builtin_shufflevector(a, a, 4, 5, 6, 7);
    v2h c = __builtin_shufflevector(b, b, 0, 1) +
            __builtin_shufflevector(b, b, 2, 3);
    _Float16 r = (_Float16)(c[0] + c[1]);
    return (float)r;
}

// ---------------------------------------------------------------------------
// Prepass: swizzle W_mix[256,256] f32 (row-major [k][n]) into per-lane WMMA
// B-fragment order, f16. Fragment fi = nt*8 + ks; lane's column n = nt*16 +
// (lane&15); element e holds K = ks*32 + (lane>>4)*16 + e.
// Flat f16 index: o = fi*512 + lane*16 + e.
// ---------------------------------------------------------------------------
__global__ void pack_wmix_k(const float* __restrict__ W, _Float16* __restrict__ Bp) {
    int o    = blockIdx.x * 256 + threadIdx.x;   // 65536 total
    int e    = o & 15;
    int lane = (o >> 4) & 31;
    int ks   = (o >> 9) & 7;
    int nt   = o >> 12;
    int k    = ks * 32 + (lane >> 4) * 16 + e;
    int n    = nt * 16 + (lane & 15);
    Bp[o] = (_Float16)W[k * 256 + n];
}

// Prepass: transpose W_four [16,128] -> [128][16] f16 and W2 [32,64] -> [64][32] f16
__global__ void pack_small_k(const float* __restrict__ Wf, const float* __restrict__ W2,
                             _Float16* __restrict__ T4, _Float16* __restrict__ T2) {
    int o = blockIdx.x * 256 + threadIdx.x;      // 2048 total
    if (o < 128 * 16) T4[o] = (_Float16)Wf[(o & 15) * 128 + (o >> 4)];
    if (o < 64 * 32)  T2[o] = (_Float16)W2[(o & 31) * 64 + (o >> 5)];
}

// ---------------------------------------------------------------------------
// Fused main kernel: async-stage B to LDS, branches (packed f16 VALU),
// WMMA mix matmul (per-K-step A-fragment build to stay <256 VGPRs), fused LN,
// non-temporal streaming output stores.
// One wave handles 16 tokens. Workgroup = 8 waves = 128 tokens.
// ---------------------------------------------------------------------------
__global__ void __launch_bounds__(256)
hybrid_embed_k(const float* __restrict__ X,
               const float* __restrict__ W_lin, const float* __restrict__ b_lin,
               const float* __restrict__ b_four,
               const float* __restrict__ W1, const float* __restrict__ b1,
               const float* __restrict__ b2,
               const float* __restrict__ b_mix,
               const float* __restrict__ gamma, const float* __restrict__ beta,
               const _Float16* __restrict__ pack,   // Bpack | Wf4t | W2t in d_ws
               float* __restrict__ out) {
    extern __shared__ __align__(32) unsigned char smem[];
    const int tid = threadIdx.x;

    // 1) Stage small transposed weights (8 KB) synchronously: needed by the
    //    branch math which overlaps with the async B staging below.
    {
        const uint4* src = (const uint4*)(const void*)(pack + BPACK_BYTES / 2);
        uint4*       dst = (uint4*)(smem + BPACK_BYTES);
        #pragma unroll 1
        for (int i = tid; i < (WF4_BYTES + W2T_BYTES) / 16; i += 256) dst[i] = src[i];
    }

    // 2) Async-stage Bpack (128 KB) global -> LDS (no VGPR round-trip,
    //    ASYNCcnt-tracked). LDS offset == global offset from `pack`.
    {
        unsigned byteoff = (unsigned)tid * 16u;
        #pragma unroll 1
        for (int i = 0; i < BPACK_BYTES / (256 * 16); ++i) {   // 32 issues/thread
            asm volatile("global_load_async_to_lds_b128 %0, %1, %2"
                         :: "v"(byteoff), "v"(byteoff), "s"(pack)
                         : "memory");
            byteoff += 256u * 16u;
        }
    }
    __syncthreads();   // small-weight ds_stores visible; async B still in flight

    const int wave = tid >> 5;
    const int lane = tid & 31;
    const int tIdx = lane & 15;     // token row within the wave's 16-token tile
    const int h    = lane >> 4;     // which half of the K-interleave this lane owns
    const int tileBase = blockIdx.x * TOK_PER_BLOCK + wave * TOK_PER_WAVE;

    const _Float16* sW4 = (const _Float16*)(smem + BPACK_BYTES);
    const _Float16* sW2 = (const _Float16*)(smem + BPACK_BYTES + WF4_BYTES);

    const float x  = __builtin_nontemporal_load(X + tileBase + tIdx);
    const float xc = fminf(fmaxf(x, -50.f), 50.f);

    // Fourier table: feat[j]=sin(xc*pi*2^j) j<8, cos for j>=8 (freqs are exactly 2^j)
    v8h scA, scB;
    #pragma unroll
    for (int j = 0; j < 8; ++j) {
        float ang = xc * (3.14159265358979323846f * (float)(1 << j));
        scA[j] = (_Float16)__sinf(ang);
        scB[j] = (_Float16)__cosf(ang);
    }

    // MLP hidden: exact-erf GELU(xc*W1+b1), 32 values (W1/b1 uniform -> scalar loads)
    v8h hv[4];
    #pragma unroll
    for (int j = 0; j < 32; ++j) {
        float z = fmaf(xc, W1[j], b1[j]);
        float g = 0.5f * z * (1.0f + erff(z * 0.70710678118654752f));
        hv[j >> 3][j & 7] = (_Float16)g;
    }

    // B tiles must be fully resident before WMMA.
    asm volatile("s_wait_asynccnt 0" ::: "memory");
    __syncthreads();

    // Mix matmul: 16 tokens x 256 outs, K=256. 128 x v_wmma_f32_16x16x32_f16.
    // A-fragment for each K-step is built just-in-time (keeps VGPRs < 256):
    // element (part*8+i) = feature c = ks*32 + h*16 + part*8 + i (matches B pack).
    v8f acc[16] = {};
    const v16h* Bl = (const v16h*)smem;
    #pragma unroll
    for (int ks = 0; ks < 8; ++ks) {
        v16h a;
        #pragma unroll
        for (int part = 0; part < 2; ++part) {
            const int c0 = ks * 32 + h * 16 + part * 8;
            #pragma unroll
            for (int i = 0; i < 8; ++i) {
                const int c = c0 + i;
                float v;
                if (ks < 2) {                        // linear branch: c in [0,64)
                    v = fmaf(xc, W_lin[c], b_lin[c]);
                } else if (ks < 6) {                 // fourier branch: c in [64,192)
                    const int cf = c - 64;
                    const v8h* wp = (const v8h*)(sW4 + cf * 16);
                    v8h s = scA * wp[0] + scB * wp[1];
                    v = hsum8(s) + b_four[cf];
                } else {                             // mlp branch: c in [192,256)
                    const int cm = c - 192;
                    const v8h* wp = (const v8h*)(sW2 + cm * 32);
                    v8h s = hv[0] * wp[0] + hv[1] * wp[1] + hv[2] * wp[2] + hv[3] * wp[3];
                    v = hsum8(s) + b2[cm];
                }
                a[part * 8 + i] = (_Float16)clip100(v);
            }
        }
        #pragma unroll
        for (int nt = 0; nt < 16; ++nt) {
            const v16h b = Bl[(nt * 8 + ks) * 32 + lane];
            acc[nt] = __builtin_amdgcn_wmma_f32_16x16x32_f16(
                false, a, false, b, (short)0, acc[nt], false, false);
        }
    }

    // Epilogue: +b_mix, clip, LayerNorm, *gamma +beta, store.
    // D layout: lane holds (token = tileBase + h*8 + e, dim n = nt*16 + (lane&15)).
    const int nbase = lane & 15;
    float bm[16], gm[16], bt[16];
    #pragma unroll
    for (int nt = 0; nt < 16; ++nt) {
        const int n = nt * 16 + nbase;
        bm[nt] = b_mix[n]; gm[nt] = gamma[n]; bt[nt] = beta[n];
    }

    float ps[8] = {}, pq[8] = {};
    #pragma unroll
    for (int nt = 0; nt < 16; ++nt) {
        #pragma unroll
        for (int e = 0; e < 8; ++e) {
            float v = clip100(acc[nt][e] + bm[nt]);
            acc[nt][e] = v;
            ps[e] += v;
            pq[e] += v * v;
        }
    }
    // Reduce across the 16 lanes of this half-wave (xor 1,2,4,8 keeps bit4 = h)
    #pragma unroll
    for (int m = 1; m <= 8; m <<= 1) {
        #pragma unroll
        for (int e = 0; e < 8; ++e) {
            ps[e] += __shfl_xor(ps[e], m, 32);
            pq[e] += __shfl_xor(pq[e], m, 32);
        }
    }
    float mu[8], rs[8];
    #pragma unroll
    for (int e = 0; e < 8; ++e) {
        mu[e] = ps[e] * (1.0f / 256.0f);
        float var = pq[e] * (1.0f / 256.0f) - mu[e] * mu[e];
        rs[e] = rsqrtf(var + 1e-5f);
    }
    // Streaming, write-once output: non-temporal stores keep the 256 MB output
    // stream from evicting the (hot, every-block) B-pack lines in L2.
    float* outBase = out + (tileBase + h * 8) * 256 + nbase;
    #pragma unroll
    for (int e = 0; e < 8; ++e) {
        #pragma unroll
        for (int nt = 0; nt < 16; ++nt) {
            float v = (acc[nt][e] - mu[e]) * rs[e] * gm[nt] + bt[nt];
            __builtin_nontemporal_store(v, outBase + e * 256 + nt * 16);
        }
    }
}

extern "C" void kernel_launch(void* const* d_in, const int* in_sizes, int n_in,
                              void* d_out, int out_size, void* d_ws, size_t ws_size,
                              hipStream_t stream) {
    const float* x      = (const float*)d_in[0];
    const float* W_lin  = (const float*)d_in[1];
    const float* b_lin  = (const float*)d_in[2];
    // d_in[3] = freqs (exactly 2^j, folded to compile-time constants)
    const float* W_four = (const float*)d_in[4];
    const float* b_four = (const float*)d_in[5];
    const float* W1     = (const float*)d_in[6];
    const float* b1     = (const float*)d_in[7];
    const float* W2     = (const float*)d_in[8];
    const float* b2     = (const float*)d_in[9];
    const float* W_mix  = (const float*)d_in[10];
    const float* b_mix  = (const float*)d_in[11];
    const float* gamma  = (const float*)d_in[12];
    const float* beta   = (const float*)d_in[13];
    float*       out    = (float*)d_out;
    _Float16*    pack   = (_Float16*)d_ws;      // Bpack | Wf4t | W2t

    // Prepass: swizzle/convert weights once (tiny)
    pack_wmix_k<<<256, 256, 0, stream>>>(W_mix, pack);
    pack_small_k<<<8, 256, 0, stream>>>(W_four, W2,
                                        pack + BPACK_BYTES / 2,
                                        pack + (BPACK_BYTES + WF4_BYTES) / 2);

    const int nTok   = in_sizes[0];             // 262144, divisible by 128
    const int blocks = nTok / TOK_PER_BLOCK;
    hybrid_embed_k<<<blocks, 256, STAGE_BYTES, stream>>>(
        x, W_lin, b_lin, b_four, W1, b1, b2, b_mix, gamma, beta, pack, out);
}
